// TransformerBlock_84602265796860
// MI455X (gfx1250) — matrix operand order repaired and checked by reading, not compiled, not run on hardware
//
#include <hip/hip_runtime.h>
#include <hip/hip_bf16.h>
#include <stdint.h>

// ---------------- constants (match reference dims) ----------------
constexpr int Bb  = 4;
constexpr int Ss  = 2048;
constexpr int Dd  = 1024;
constexpr int Hh  = 16;
constexpr int HKVc = 4;
constexpr int DHc = 64;
constexpr int DFFc = 4096;
constexpr int Kc  = 1024;   // MoD capacity

#define DEV __device__ __forceinline__

typedef __bf16 bf16_t;
typedef __attribute__((ext_vector_type(16))) __bf16 v16bf;
typedef __attribute__((ext_vector_type(8)))  __bf16 v8bf;
typedef __attribute__((ext_vector_type(8)))  float  v8f;
typedef __attribute__((ext_vector_type(4)))  unsigned int v4u;
typedef __attribute__((ext_vector_type(8)))  int v8i;
typedef __attribute__((ext_vector_type(4)))  int v4i;

union V16U { v16bf v; struct { v8bf lo, hi; } p; };

DEV bf16_t f2bf(float f) {
  union { float f; uint32_t u; } a; a.f = f;
  uint32_t r = a.u + 0x7fffu + ((a.u >> 16) & 1u);   // round-to-nearest-even
  union { uint16_t s; bf16_t b; } o; o.s = (uint16_t)(r >> 16);
  return o.b;
}

// A-operand fragment (16x32 bf16): lane<16 holds K {k0+0..7, k0+16..23},
// lane>=16 holds K {k0+8..15, k0+24..31} of row (passed via rowp).
DEV v16bf load_fragA(const bf16_t* rowp, int k0, int hi8) {
  const bf16_t* p = rowp + k0 + hi8;       // hi8 = (lane&16)?8:0
  V16U u; u.p.lo = *(const v8bf*)p; u.p.hi = *(const v8bf*)(p + 16);
  return u.v;
}
// B-operand fragment (32x16 bf16, K-major per column): lane<16 holds
// K {k0+0..7, k0+16..23} of column n=lane, lane>=16 holds K {k0+8..15, k0+24..31}.
DEV v16bf load_fragB(const bf16_t* colp, int k0, int lane) {
  const bf16_t* p = colp + k0 + ((lane & 16) ? 8 : 0);
  V16U u; u.p.lo = *(const v8bf*)p; u.p.hi = *(const v8bf*)(p + 16);
  return u.v;
}
DEV v8f wmma_bf16(v16bf a, v16bf b, v8f c) {
  return __builtin_amdgcn_wmma_f32_16x16x32_bf16(false, a, false, b, (short)0, c,
                                                 false, false);
}

DEV float gelu_tanh(float x) {
  const float k0 = 0.7978845608028654f;    // sqrt(2/pi)
  return 0.5f * x * (1.0f + tanhf(k0 * (x + 0.044715f * x * x * x)));
}

// ---- Tensor Data Mover: async 2-D bf16 tile load global -> LDS -------------
// D# per CDNA5 ISA ch.8: group0 {count, lds_addr, global_addr, type=2},
// group1 {data_size=2B, pad cfg, tensor dims, tile dims, dim0 stride}.
// Tile: tile_k elems contiguous (dim0) x rows (dim1), global row stride
// stride_e elems. LDS gets 16B of padding after every 128B stored
// (pad_interval=4 -> 128B, pad_amount=3 -> 4 DWORDs), i.e. padded row
// stride = tile_k*2 + 16 bytes.
DEV void tdm_load_b_tile(unsigned lds_off, const bf16_t* gptr,
                         unsigned tile_k, unsigned rows,
                         unsigned long long stride_e) {
  const unsigned long long ga = (unsigned long long)(uintptr_t)gptr;
  v4u g0 = {};
  g0.x = 1u;                                            // count=1, user D#
  g0.y = lds_off;                                       // lds_addr (bytes)
  g0.z = (unsigned)ga;                                  // global_addr[31:0]
  g0.w = (unsigned)((ga >> 32) & 0x1ffffffu) | 0x80000000u;  // [56:32] | type=2
  const unsigned td0 = 0x7fffffffu, td1 = 0x7fffffffu;  // huge: no OOB clamp
  v8i g1 = {};
  g1[0] = (int)((1u << 16) | (1u << 20) | (4u << 22) | (3u << 25));
  //           data_size=2B | pad_en   | pad_intvl=128B | pad_amt=4 DWORDs
  g1[1] = (int)((td0 & 0xffffu) << 16);                 // abar=0 | td0.lo16
  g1[2] = (int)((td0 >> 16) | ((td1 & 0xffffu) << 16)); // td0.hi16 | td1.lo16
  g1[3] = (int)((td1 >> 16) | (tile_k << 16));          // td1.hi16 | tile_dim0
  g1[4] = (int)(rows & 0xffffu);                        // tile_dim1 | tile_dim2=0
  const unsigned long long sb = stride_e;               // dim0 stride (elems)
  g1[5] = (int)(unsigned)(sb & 0xffffffffu);
  g1[6] = (int)(unsigned)((sb >> 32) & 0xffffu);        // | dim1 stride lo16 = 0
  g1[7] = 0;
  v4i g2 = {}, g3 = {};
#if defined(__clang_major__) && (__clang_major__ >= 23)
  v8i g4 = {};
  __builtin_amdgcn_tensor_load_to_lds(g0, g1, g2, g3, g4, 0);
#else
  __builtin_amdgcn_tensor_load_to_lds(g0, g1, g2, g3, 0);
#endif
}

// ---------------- weight convert + transpose: wt[n][k] = bf16(w[k][n]) ------
__global__ __launch_bounds__(256)
void convertT_kernel(const float* __restrict__ w, bf16_t* __restrict__ wt,
                     int Kd, int N) {
  const int n = blockIdx.x * 32 + (threadIdx.x & 31);
  const int k = blockIdx.y * 8 + (threadIdx.x >> 5);
  wt[(size_t)n * Kd + k] = f2bf(w[(size_t)k * N + n]);
}

// ---------------- fused RMSNorm + router score ------------------------------
__global__ __launch_bounds__(256)
void rmsnorm_score_kernel(const float* __restrict__ x, const float* __restrict__ w,
                          const float* __restrict__ rw, bf16_t* __restrict__ hout,
                          float* __restrict__ scores) {
  __shared__ float red[256];
  const size_t row = blockIdx.x;
  const int tid = threadIdx.x;
  const float* xr = x + row * Dd;
  float v0 = xr[tid], v1 = xr[tid + 256], v2 = xr[tid + 512], v3 = xr[tid + 768];
  red[tid] = v0 * v0 + v1 * v1 + v2 * v2 + v3 * v3;
  __syncthreads();
  for (int s = 128; s > 0; s >>= 1) { if (tid < s) red[tid] += red[tid + s]; __syncthreads(); }
  const float scale = rsqrtf(red[0] * (1.0f / Dd) + 1e-6f);
  __syncthreads();
  const float y0 = v0 * scale * w[tid],       y1 = v1 * scale * w[tid + 256];
  const float y2 = v2 * scale * w[tid + 512], y3 = v3 * scale * w[tid + 768];
  bf16_t* ho = hout + row * Dd;
  ho[tid] = f2bf(y0); ho[tid + 256] = f2bf(y1);
  ho[tid + 512] = f2bf(y2); ho[tid + 768] = f2bf(y3);
  red[tid] = y0 * rw[tid] + y1 * rw[tid + 256] + y2 * rw[tid + 512] + y3 * rw[tid + 768];
  __syncthreads();
  for (int s = 128; s > 0; s >>= 1) { if (tid < s) red[tid] += red[tid + s]; __syncthreads(); }
  if (tid == 0) scores[row] = red[0];
}

// ---------------- exact top-K (rank counting, lax.top_k tie-break) ----------
__global__ __launch_bounds__(256)
void topk_kernel(const float* __restrict__ scores, int* __restrict__ idx) {
  __shared__ float sc[Ss];
  __shared__ unsigned char fl[Ss];
  const int b = blockIdx.x, tid = threadIdx.x;
  for (int i = tid; i < Ss; i += 256) sc[i] = scores[b * Ss + i];
  __syncthreads();
  for (int i = tid; i < Ss; i += 256) {
    const float si = sc[i];
    int rank = 0;
    for (int j = 0; j < Ss; ++j) {
      const float sj = sc[j];
      rank += (sj > si) || ((sj == si) && (j < i));
    }
    fl[i] = (rank < Kc) ? 1 : 0;
  }
  __syncthreads();
  for (int i = tid; i < Ss; i += 256) {
    if (fl[i]) {
      int pos = 0;
      for (int j = 0; j < i; ++j) pos += fl[j];
      idx[b * Kc + pos] = i;   // ascending since i ascending over selected
    }
  }
}

// ---------------- gather selected rows (bf16) -------------------------------
__global__ __launch_bounds__(256)
void gather_kernel(const bf16_t* __restrict__ h, const int* __restrict__ idx,
                   bf16_t* __restrict__ sel) {
  const int b = blockIdx.x >> 10;
  const int src = idx[blockIdx.x];
  const uint2* s = (const uint2*)(h + ((size_t)b * Ss + src) * Dd);
  uint2* d = (uint2*)(sel + (size_t)blockIdx.x * Dd);
  d[threadIdx.x] = s[threadIdx.x];
}

// ---------------- generic WMMA GEMM: C = A(MxKd) * Bt(NxKd)^T ---------------
// B operand streamed by the Tensor Data Mover into double-buffered LDS
// (per-wave 64xKCH bf16 tiles, padded rows -> conflict-free ds_load_b128).
// MODE 0: f32 out [b][m][n]
// MODE 1: bf16 transposed out [b][n][m]        (V -> V^T for attention)
// MODE 2: gelu -> bf16 out [b][m][n]           (FFN mid)
// MODE 3: scatter: out[b][idx[b][m]][n] = res[b][idx[b][m]][n] + acc
template <int MODE>
__global__ __launch_bounds__(128)
void gemm_bf16_kernel(const bf16_t* __restrict__ A, const bf16_t* __restrict__ Bt,
                      float* __restrict__ outF, bf16_t* __restrict__ outB,
                      const float* __restrict__ res, const int* __restrict__ idx,
                      int M, int N, int Kd, size_t strideA) {
  constexpr int KCH  = 64;                 // K-chunk per TDM transfer
  constexpr int ROWE = KCH + 8;            // padded LDS row stride (elements)
  __shared__ __align__(16) bf16_t bl[2 * 4 * 64 * ROWE];   // 72 KB
  const int lane = threadIdx.x & 31;
  const int wv   = threadIdx.x >> 5;
  const int b    = blockIdx.z;
  const int Mbase = blockIdx.x * 16;
  const int Nbase = blockIdx.y * 256 + wv * 64;
  if (Nbase >= N) return;
  const int lq  = lane & 15;
  const int hi8 = (lane & 16) ? 8 : 0;
  const bf16_t* Arow = A + (size_t)b * strideA + (size_t)(Mbase + lq) * Kd;
  const bf16_t* gB = Bt + (size_t)Nbase * Kd;

  const unsigned lbase = (unsigned)(uintptr_t)&bl[0];        // LDS byte offset
  const unsigned bufsz = (unsigned)(64 * ROWE * sizeof(bf16_t));
  const unsigned boff0 = lbase + (unsigned)wv * bufsz;
  const unsigned boff1 = lbase + (unsigned)(4 + wv) * bufsz;
  bf16_t* lds0 = &bl[(size_t)wv * 64 * ROWE];
  bf16_t* lds1 = &bl[(size_t)(4 + wv) * 64 * ROWE];

  tdm_load_b_tile(boff0, gB, KCH, 64, (unsigned long long)Kd);

  v8f acc[4] = {};
  const int nch = Kd / KCH;
  for (int c = 0; c < nch; ++c) {
    if (c + 1 < nch) {
      tdm_load_b_tile((c & 1) ? boff0 : boff1, gB + (size_t)(c + 1) * KCH, KCH, 64,
                      (unsigned long long)Kd);
      __builtin_amdgcn_s_wait_tensorcnt(1);    // chunk c landed (in-order)
    } else {
      __builtin_amdgcn_s_wait_tensorcnt(0);
    }
    asm volatile("" ::: "memory");             // keep ds reads below the wait
    const bf16_t* lw = (c & 1) ? lds1 : lds0;
#pragma unroll
    for (int kk = 0; kk < KCH; kk += 32) {
      __builtin_prefetch(Arow + c * KCH + kk + 64, 0, 1);
      const v16bf a = load_fragA(Arow, c * KCH + kk, hi8);
      acc[0] = wmma_bf16(a, load_fragB(lw + (size_t)(0 * 16 + lq) * ROWE, kk, lane), acc[0]);
      acc[1] = wmma_bf16(a, load_fragB(lw + (size_t)(1 * 16 + lq) * ROWE, kk, lane), acc[1]);
      acc[2] = wmma_bf16(a, load_fragB(lw + (size_t)(2 * 16 + lq) * ROWE, kk, lane), acc[2]);
      acc[3] = wmma_bf16(a, load_fragB(lw + (size_t)(3 * 16 + lq) * ROWE, kk, lane), acc[3]);
    }
    asm volatile("" ::: "memory");             // reads done before next reuse
  }
#pragma unroll
  for (int t = 0; t < 4; ++t) {
#pragma unroll
    for (int r = 0; r < 8; ++r) {
      const float v = acc[t][r];
      const int m = Mbase + r + hi8;
      const int n = Nbase + t * 16 + lq;
      if constexpr (MODE == 0) {
        outF[((size_t)b * M + m) * N + n] = v;
      } else if constexpr (MODE == 1) {
        outB[((size_t)b * N + n) * M + m] = f2bf(v);
      } else if constexpr (MODE == 2) {
        outB[((size_t)b * M + m) * N + n] = f2bf(gelu_tanh(v));
      } else {
        const int tok = idx[b * M + m];
        const size_t o = ((size_t)b * Ss + tok) * Dd + n;
        outF[o] = res[o] + v;
      }
    }
  }
}

// ---------------- RoPE + bf16 convert ---------------------------------------
__global__ __launch_bounds__(64)
void rope_kernel(const float* __restrict__ in, bf16_t* __restrict__ outb, int heads) {
  const int gid = blockIdx.x;                // (b*Kc + t)*heads + h
  const int d   = threadIdx.x;
  const int t   = (gid / heads) & (Kc - 1);
  const size_t off = (size_t)gid * DHc + d;
  const float x = in[off];
  const float ang = (float)t * __expf(-(float)(d & 31) * 0.28782313662425572f); // ln(1e4)/32
  const float c = __cosf(ang), s = __sinf(ang);
  const float other = (d < 32) ? -in[off + 32] : in[off - 32];
  outb[off] = f2bf(x * c + other * s);
}

// ---------------- flash-style causal GQA attention --------------------------
// grid: (Kc/16, Hh, Bb), block 32 (one wave). q:[b][t][h][64]  k:[b][t][hkv][64]
// vT:[b][hkv*64+d][t]  -> out o:[b][t][h][64] (bf16)
__global__ __launch_bounds__(32)
void attn_kernel(const bf16_t* __restrict__ qb, const bf16_t* __restrict__ kb,
                 const bf16_t* __restrict__ vt, bf16_t* __restrict__ ob) {
  __shared__ __align__(16) bf16_t pbuf[16 * 32];
  const int qt = blockIdx.x, h = blockIdx.y, b = blockIdx.z;
  const int lane = threadIdx.x;
  const int lq  = lane & 15;
  const int hi8 = (lane & 16) ? 8 : 0;
  const int hkv = h >> 2;                    // rep = H/HKV = 4

  const bf16_t* qrow = qb + (((size_t)(b * Kc + qt * 16 + lq)) * Hh + h) * DHc;
  const v16bf qf0 = load_fragA(qrow, 0, hi8);
  const v16bf qf1 = load_fragA(qrow, 32, hi8);

  const bf16_t* kbase = kb + (size_t)b * Kc * HKVc * DHc + hkv * DHc;
  const bf16_t* vbase = vt + ((size_t)b * (HKVc * DHc) + hkv * DHc) * Kc;

  v8f o[4] = {};
  float mrun[8], lrun[8];
#pragma unroll
  for (int r = 0; r < 8; ++r) { mrun[r] = -3.0e38f; lrun[r] = 0.f; }

  const int nb = (qt >> 1) + 1;              // 32-key blocks up to diagonal
  for (int blk = 0; blk < nb; ++blk) {
    const int kb0 = blk * 32;
    v8f sc0 = {}, sc1 = {};
    {
      const bf16_t* kr0 = kbase + (size_t)(kb0 + lq) * (HKVc * DHc);
      const bf16_t* kr1 = kbase + (size_t)(kb0 + 16 + lq) * (HKVc * DHc);
      sc0 = wmma_bf16(qf0, load_fragB(kr0, 0, lane), sc0);
      sc0 = wmma_bf16(qf1, load_fragB(kr0, 32, lane), sc0);
      sc1 = wmma_bf16(qf0, load_fragB(kr1, 0, lane), sc1);
      sc1 = wmma_bf16(qf1, load_fragB(kr1, 32, lane), sc1);
    }
#pragma unroll
    for (int r = 0; r < 8; ++r) {
      const int qi = qt * 16 + r + hi8;
      float x0 = sc0[r] * 0.125f;            // 1/sqrt(64)
      float x1 = sc1[r] * 0.125f;
      if (kb0 + lq > qi)      x0 = -1.0e30f;
      if (kb0 + 16 + lq > qi) x1 = -1.0e30f;
      float rm = fmaxf(x0, x1);
#pragma unroll
      for (int m = 1; m < 16; m <<= 1) rm = fmaxf(rm, __shfl_xor(rm, m, 32));
      const float nm = fmaxf(mrun[r], rm);
      const float al = __expf(mrun[r] - nm);
      const float p0 = __expf(x0 - nm);
      const float p1 = __expf(x1 - nm);
      float rs = p0 + p1;
#pragma unroll
      for (int m = 1; m < 16; m <<= 1) rs += __shfl_xor(rs, m, 32);
      lrun[r] = lrun[r] * al + rs;
      mrun[r] = nm;
      o[0][r] *= al; o[1][r] *= al; o[2][r] *= al; o[3][r] *= al;
      pbuf[(r + hi8) * 32 + lq]      = f2bf(p0);   // C-layout -> (m,k) in LDS
      pbuf[(r + hi8) * 32 + 16 + lq] = f2bf(p1);
    }
    const v16bf pa = load_fragA(&pbuf[lq * 32], 0, hi8);   // P as A-operand
#pragma unroll
    for (int t = 0; t < 4; ++t) {
      const bf16_t* vcol = vbase + (size_t)(t * 16 + lq) * Kc;
      o[t] = wmma_bf16(pa, load_fragB(vcol, kb0, lane), o[t]);
    }
  }
#pragma unroll
  for (int t = 0; t < 4; ++t) {
#pragma unroll
    for (int r = 0; r < 8; ++r) {
      const float val = o[t][r] / lrun[r];
      const int tok = qt * 16 + r + hi8;
      ob[(((size_t)(b * Kc + tok)) * Hh + h) * DHc + t * 16 + lq] = f2bf(val);
    }
  }
}

// ---------------- out = 2*in (residual + residual for unselected rows) -----
__global__ __launch_bounds__(256)
void scale2_kernel(const float* __restrict__ in, float* __restrict__ outp) {
  const size_t i = (size_t)blockIdx.x * 256 + threadIdx.x;
  float4 v = ((const float4*)in)[i];
  v.x *= 2.f; v.y *= 2.f; v.z *= 2.f; v.w *= 2.f;
  ((float4*)outp)[i] = v;
}

// ---------------- host-side launch ------------------------------------------
extern "C" void kernel_launch(void* const* d_in, const int* in_sizes, int n_in,
                              void* d_out, int out_size, void* d_ws, size_t ws_size,
                              hipStream_t stream) {
  (void)in_sizes; (void)n_in; (void)out_size; (void)ws_size;
  const float* hs  = (const float*)d_in[0];
  const float* snw = (const float*)d_in[1];
  const float* fnw = (const float*)d_in[2];
  const float* wq  = (const float*)d_in[3];
  const float* wk  = (const float*)d_in[4];
  const float* wv  = (const float*)d_in[5];
  const float* wo  = (const float*)d_in[6];
  const float* srw = (const float*)d_in[7];
  const float* frw = (const float*)d_in[8];
  const float* w1  = (const float*)d_in[9];
  const float* w2  = (const float*)d_in[10];
  float* out = (float*)d_out;

  char* ws = (char*)d_ws;
  size_t off = 0;
  auto alloc = [&](size_t bytes) {
    char* p = ws + off; off += (bytes + 255) & ~(size_t)255; return p;
  };
  bf16_t* wqT   = (bf16_t*)alloc((size_t)Dd * (Hh * DHc) * 2);
  bf16_t* wkT   = (bf16_t*)alloc((size_t)Dd * (HKVc * DHc) * 2);
  bf16_t* wvT   = (bf16_t*)alloc((size_t)Dd * (HKVc * DHc) * 2);
  bf16_t* woT   = (bf16_t*)alloc((size_t)(Hh * DHc) * Dd * 2);
  bf16_t* w1T   = (bf16_t*)alloc((size_t)Dd * DFFc * 2);
  bf16_t* w2T   = (bf16_t*)alloc((size_t)DFFc * Dd * 2);
  bf16_t* hbf   = (bf16_t*)alloc((size_t)Bb * Ss * Dd * 2);
  float*  scores = (float*)alloc((size_t)Bb * Ss * 4);
  int*    idx1  = (int*)alloc((size_t)Bb * Kc * 4);
  int*    idx2  = (int*)alloc((size_t)Bb * Kc * 4);
  bf16_t* selbf = (bf16_t*)alloc((size_t)Bb * Kc * Dd * 2);
  // Region reused for FFN mid (32MB): qf32(16)+kf32(4)+qbf(8)+kbf(2)+vT(2)
  char* reuse_base = ws + off;
  float*  qf32  = (float*)alloc((size_t)Bb * Kc * Hh * DHc * 4);
  float*  kf32  = (float*)alloc((size_t)Bb * Kc * HKVc * DHc * 4);
  bf16_t* qbf   = (bf16_t*)alloc((size_t)Bb * Kc * Hh * DHc * 2);
  bf16_t* kbf   = (bf16_t*)alloc((size_t)Bb * Kc * HKVc * DHc * 2);
  bf16_t* vTbf  = (bf16_t*)alloc((size_t)Bb * HKVc * DHc * Kc * 2);
  bf16_t* midbf = (bf16_t*)reuse_base;          // B*K*DFF bf16, aliases the above
  bf16_t* obf   = (bf16_t*)alloc((size_t)Bb * Kc * Hh * DHc * 2);
  float*  hidden = (float*)alloc((size_t)Bb * Ss * Dd * 4);

  const dim3 b256(256), b128(128), b64(64), b32(32);

  // --- weight convert + transpose (bf16, [N][K] layout) ---
  convertT_kernel<<<dim3(1024 / 32, 1024 / 8), b256, 0, stream>>>(wq, wqT, 1024, 1024);
  convertT_kernel<<<dim3(256 / 32, 1024 / 8),  b256, 0, stream>>>(wk, wkT, 1024, 256);
  convertT_kernel<<<dim3(256 / 32, 1024 / 8),  b256, 0, stream>>>(wv, wvT, 1024, 256);
  convertT_kernel<<<dim3(1024 / 32, 1024 / 8), b256, 0, stream>>>(wo, woT, 1024, 1024);
  convertT_kernel<<<dim3(4096 / 32, 1024 / 8), b256, 0, stream>>>(w1, w1T, 1024, 4096);
  convertT_kernel<<<dim3(1024 / 32, 4096 / 8), b256, 0, stream>>>(w2, w2T, 4096, 1024);

  // ================= attention sub-layer =================
  rmsnorm_score_kernel<<<dim3(Bb * Ss), b256, 0, stream>>>(hs, snw, srw, hbf, scores);
  topk_kernel<<<dim3(Bb), b256, 0, stream>>>(scores, idx1);
  gather_kernel<<<dim3(Bb * Kc), b256, 0, stream>>>(hbf, idx1, selbf);

  // q,k,v projections (per-batch 1024x{1024,256,256} GEMMs, Kd=1024)
  gemm_bf16_kernel<0><<<dim3(Kc / 16, 4, Bb), b128, 0, stream>>>(
      selbf, wqT, qf32, nullptr, nullptr, nullptr, Kc, 1024, Dd, (size_t)Kc * Dd);
  gemm_bf16_kernel<0><<<dim3(Kc / 16, 1, Bb), b128, 0, stream>>>(
      selbf, wkT, kf32, nullptr, nullptr, nullptr, Kc, 256, Dd, (size_t)Kc * Dd);
  gemm_bf16_kernel<1><<<dim3(Kc / 16, 1, Bb), b128, 0, stream>>>(
      selbf, wvT, nullptr, vTbf, nullptr, nullptr, Kc, 256, Dd, (size_t)Kc * Dd);

  rope_kernel<<<dim3(Bb * Kc * Hh), b64, 0, stream>>>(qf32, qbf, Hh);
  rope_kernel<<<dim3(Bb * Kc * HKVc), b64, 0, stream>>>(kf32, kbf, HKVc);

  attn_kernel<<<dim3(Kc / 16, Hh, Bb), b32, 0, stream>>>(qbf, kbf, vTbf, obf);

  // hidden = 2*residual everywhere, then selected rows = residual + attn@wo
  scale2_kernel<<<dim3((Bb * Ss * Dd) / 4 / 256), b256, 0, stream>>>(hs, hidden);
  gemm_bf16_kernel<3><<<dim3(Kc / 16, 4, Bb), b128, 0, stream>>>(
      obf, woT, hidden, nullptr, hs, idx1, Kc, 1024, Hh * DHc, (size_t)Kc * Hh * DHc);

  // ================= FFN sub-layer =================
  rmsnorm_score_kernel<<<dim3(Bb * Ss), b256, 0, stream>>>(hidden, fnw, frw, hbf, scores);
  topk_kernel<<<dim3(Bb), b256, 0, stream>>>(scores, idx2);
  gather_kernel<<<dim3(Bb * Kc), b256, 0, stream>>>(hbf, idx2, selbf);

  gemm_bf16_kernel<2><<<dim3(Kc / 16, DFFc / 256, Bb), b128, 0, stream>>>(
      selbf, w1T, nullptr, midbf, nullptr, nullptr, Kc, DFFc, Dd, (size_t)Kc * Dd);

  scale2_kernel<<<dim3((Bb * Ss * Dd) / 4 / 256), b256, 0, stream>>>(hidden, out);
  gemm_bf16_kernel<3><<<dim3(Kc / 16, 4, Bb), b128, 0, stream>>>(
      midbf, w2T, out, nullptr, hidden, idx2, Kc, 1024, DFFc, (size_t)Kc * DFFc);
}
